// embedded_label_wise_attention_90400471646330
// MI455X (gfx1250) — matrix operand order repaired and checked by reading, not compiled, not run on hardware
//
#include <hip/hip_runtime.h>
#include <stdint.h>

#define B_ 16
#define S_ 4096
#define D_ 256
#define C_ 512

typedef __bf16 bf16;
typedef __attribute__((ext_vector_type(16))) __bf16 v16bf;
typedef __attribute__((ext_vector_type(8)))  __bf16 v8bf;
typedef __attribute__((ext_vector_type(4)))  __bf16 v4bf;
typedef __attribute__((ext_vector_type(8)))  float  v8f;
typedef __attribute__((ext_vector_type(4)))  float  v4f;

static __device__ __forceinline__ v16bf cat16(v8bf lo, v8bf hi) {
  return __builtin_shufflevector(lo, hi, 0,1,2,3,4,5,6,7,8,9,10,11,12,13,14,15);
}

__launch_bounds__(256, 1)
__global__ void elwa_kernel(const float* __restrict__ xg,
                            const float* __restrict__ wg,
                            const float* __restrict__ bg,
                            float* __restrict__ outg)
{
  // 128 KB staging region, aliased:
  //   during loop : sXr [4][32][256] bf16 (64KB) + sXt [4][256][32] bf16 (64KB)
  //   at start    : tmpW [32][256] f32 (32KB)
  //   after loop  : sComb [32][256] f32 (32KB)
  __shared__ __align__(16) unsigned char smem[131072];
  __shared__ __align__(16) bf16 sW[32 * 256];     // W block, bf16, row-major (c,d) (16KB)
  __shared__ __align__(16) bf16 sP[8][16 * 32];   // per-wave P tile (c,s) (8KB)
  __shared__ float sM[2][4][16];
  __shared__ float sL[2][4][16];
  __shared__ float sInvL[32];

  bf16*  sXr   = (bf16*)smem;             // row-major  (s, d)
  bf16*  sXt   = (bf16*)(smem + 65536);   // transposed (d, s)
  float* sComb = (float*)smem;

  const int tid   = threadIdx.x;
  const int wave  = tid >> 5;
  const int lane  = tid & 31;
  const int ln    = lane & 15;   // lane within half
  const int lh    = lane >> 4;   // lane half (0/1)
  const int ct    = wave >> 2;   // c-tile within block: 0..1
  const int ssp   = wave & 3;    // s-split: 0..3
  const int cblk  = blockIdx.x * 32;
  const int batch = blockIdx.y;

  const float* xb = xg + (size_t)batch * (S_ * D_);

  // ---------- stage W(32x256 f32) -> LDS via async copy, convert to bf16 ----------
  {
    float* tmpW = (float*)smem;
    const uint32_t ldsbase = (uint32_t)(uintptr_t)tmpW;
    const uint64_t gbase   = (uint64_t)(uintptr_t)(wg + (size_t)cblk * D_);
    #pragma unroll
    for (int k = 0; k < 8; ++k) {
      const int inst = wave * 8 + k;
      uint32_t loff = ldsbase + inst * 512 + lane * 16;
      uint64_t ga   = gbase   + (uint64_t)(inst * 512 + lane * 16);
      asm volatile("global_load_async_to_lds_b128 %0, %1, off"
                   :: "v"(loff), "v"(ga) : "memory");
    }
    asm volatile("s_wait_asynccnt 0" ::: "memory");
    __syncthreads();
    #pragma unroll
    for (int j = 0; j < 32; ++j) {
      const int idx = tid * 32 + j;
      sW[idx] = (bf16)tmpW[idx];
    }
    __syncthreads();
  }

  // ---------- flash state ----------
  v8f acc[16];                       // 16 d-tiles of 16x16 f32 (rows=c, cols=d)
  #pragma unroll
  for (int i = 0; i < 16; ++i)
    #pragma unroll
    for (int v = 0; v < 8; ++v) acc[i][v] = 0.0f;

  float mc = -1e30f;                 // running max for column c = cblk+ct*16+ln
  float lc = 0.0f;                   // running sum for that column

  // staging decomposition: 64 threads per s-split buffer
  const int sbuf = tid >> 6;
  const int u6   = tid & 63;
  const int sg   = u6 >> 3;          // s group of 4 rows
  const int dg   = u6 & 7;           // d group of 32 cols

  for (int it = 0; it < 32; ++it) {
    // ---- cooperative staging: x chunk (32 s x 256 d) per s-split, f32 -> bf16, 2 layouts ----
    {
      const int s0 = sbuf * 1024 + it * 32;
      bf16* xr = sXr + sbuf * (32 * 256);
      bf16* xt = sXt + sbuf * (256 * 32);
      #pragma unroll
      for (int db = 0; db < 4; ++db) {
        const int d0 = dg * 32 + db * 8;
        float f[4][8];
        #pragma unroll
        for (int si = 0; si < 4; ++si) {
          const float* src = xb + (size_t)(s0 + sg * 4 + si) * D_ + d0;
          v4f a  = *(const v4f*)src;
          v4f b2 = *(const v4f*)(src + 4);
          f[si][0] = a[0];  f[si][1] = a[1];  f[si][2] = a[2];  f[si][3] = a[3];
          f[si][4] = b2[0]; f[si][5] = b2[1]; f[si][6] = b2[2]; f[si][7] = b2[3];
        }
        #pragma unroll
        for (int si = 0; si < 4; ++si) {
          v8bf h;
          #pragma unroll
          for (int k = 0; k < 8; ++k) h[k] = (bf16)f[si][k];
          *(v8bf*)&xr[(sg * 4 + si) * 256 + d0] = h;
        }
        #pragma unroll
        for (int k = 0; k < 8; ++k) {
          v4bf t;
          #pragma unroll
          for (int si = 0; si < 4; ++si) t[si] = (bf16)f[si][k];
          *(v4bf*)&xt[(d0 + k) * 32 + sg * 4] = t;
        }
      }
      // prefetch next chunk toward the caches (global_prefetch_b8)
      if (it + 1 < 32) {
        const int s0n = sbuf * 1024 + (it + 1) * 32;
        #pragma unroll
        for (int si = 0; si < 4; ++si)
          __builtin_prefetch(xb + (size_t)(s0n + sg * 4 + si) * D_ + dg * 32, 0, 1);
      }
    }
    __syncthreads();

    const int  scb = ssp * 1024 + it * 32;
    const bf16* xr = sXr + ssp * (32 * 256);
    const bf16* xt = sXt + ssp * (256 * 32);

    // ---- GEMM1 (transposed): scores^T(32s x 16c) = x_chunk x W_tile^T ----
    //      A = x (M=s, K=d) from row-major sXr; B = W (K=d, N=c) from row-major sW
    v8f sc0, sc1;  // st=0: s_rel 0..15, st=1: s_rel 16..31 ; rows=s, cols=c
    #pragma unroll
    for (int v = 0; v < 8; ++v) { sc0[v] = 0.0f; sc1[v] = 0.0f; }

    #pragma unroll
    for (int kc = 0; kc < 8; ++kc) {
      const int kb = kc * 32;
      const bf16* wr  = &sW[(ct * 16 + ln) * 256 + kb];
      v16bf bw = cat16(*(const v8bf*)&wr[lh * 8], *(const v8bf*)&wr[16 + lh * 8]);
      const bf16* xa0 = &xr[ln * 256 + kb];
      const bf16* xa1 = &xr[(16 + ln) * 256 + kb];
      v16bf a0 = cat16(*(const v8bf*)&xa0[lh * 8], *(const v8bf*)&xa0[16 + lh * 8]);
      v16bf a1 = cat16(*(const v8bf*)&xa1[lh * 8], *(const v8bf*)&xa1[16 + lh * 8]);
      sc0 = __builtin_amdgcn_wmma_f32_16x16x32_bf16(false, a0, false, bw, (short)0, sc0, false, false);
      sc1 = __builtin_amdgcn_wmma_f32_16x16x32_bf16(false, a1, false, bw, (short)0, sc1, false, false);
    }

    // ---- bias (vectorized per-lane) ----
    {
      const int crow = cblk + ct * 16 + ln;
      const float* bptr = bg + (size_t)crow * S_ + scb;
      v4f b0a = *(const v4f*)(bptr + 8 * lh);
      v4f b0b = *(const v4f*)(bptr + 8 * lh + 4);
      v4f b1a = *(const v4f*)(bptr + 16 + 8 * lh);
      v4f b1b = *(const v4f*)(bptr + 16 + 8 * lh + 4);
      #pragma unroll
      for (int v = 0; v < 4; ++v) {
        sc0[v]     += b0a[v];
        sc0[v + 4] += b0b[v];
        sc1[v]     += b1a[v];
        sc1[v + 4] += b1b[v];
      }
    }

    // ---- online softmax: per-lane column stats, 1 shuffle per reduction ----
    float t = -1e30f;
    #pragma unroll
    for (int v = 0; v < 8; ++v) t = fmaxf(t, fmaxf(sc0[v], sc1[v]));
    t = fmaxf(t, __shfl_xor(t, 16, 32));
    const float mn = fmaxf(mc, t);
    const float pscale = __expf(mc - mn);
    mc = mn;
    float rs = 0.0f;
    #pragma unroll
    for (int v = 0; v < 8; ++v) {
      sc0[v] = __expf(sc0[v] - mn);
      sc1[v] = __expf(sc1[v] - mn);
      rs += sc0[v] + sc1[v];
    }
    rs += __shfl_xor(rs, 16, 32);
    lc = lc * pscale + rs;

    // broadcast per-column scale to acc row slots (row r = v + 8*lh holds c_rel = r)
    float rscale[8];
    #pragma unroll
    for (int v = 0; v < 8; ++v) rscale[v] = __shfl(pscale, v + 8 * lh, 32);
    #pragma unroll
    for (int dt = 0; dt < 16; ++dt)
      #pragma unroll
      for (int v = 0; v < 8; ++v) acc[dt][v] *= rscale[v];

    // ---- P relayout through LDS: two contiguous 8-half vector stores per lane ----
    bf16* pw = sP[wave];
    {
      v8bf p0, p1;
      #pragma unroll
      for (int v = 0; v < 8; ++v) { p0[v] = (bf16)sc0[v]; p1[v] = (bf16)sc1[v]; }
      *(v8bf*)&pw[ln * 32 + 8 * lh]      = p0;   // (c=ln, s = 8*lh + v)
      *(v8bf*)&pw[ln * 32 + 16 + 8 * lh] = p1;   // (c=ln, s = 16 + 8*lh + v)
    }
    const bf16* pr = &pw[ln * 32];
    v16bf pfrag = cat16(*(const v8bf*)&pr[lh * 8], *(const v8bf*)&pr[16 + lh * 8]);

    // ---- GEMM2: acc(16c x 256d) += P(16c x 32s) x x(32s x 256d) ----
    #pragma unroll
    for (int dt = 0; dt < 16; ++dt) {
      const bf16* xc = &xt[(dt * 16 + ln) * 32];
      v16bf bfrag = cat16(*(const v8bf*)&xc[lh * 8], *(const v8bf*)&xc[16 + lh * 8]);
      acc[dt] = __builtin_amdgcn_wmma_f32_16x16x32_bf16(false, pfrag, false, bfrag, (short)0, acc[dt], false, false);
    }
    __syncthreads();
  }

  // ---------- combine the 4 s-split partials ----------
  if (lh == 0) {
    sM[ct][ssp][ln] = mc;
    sL[ct][ssp][ln] = lc;
  }
  #pragma unroll
  for (int j = 0; j < 32; ++j) sComb[tid * 32 + j] = 0.0f;
  __syncthreads();

  {
    float Mg = -1e30f;
    #pragma unroll
    for (int s2 = 0; s2 < 4; ++s2) Mg = fmaxf(Mg, sM[ct][s2][ln]);
    float Lg = 0.0f;
    #pragma unroll
    for (int s2 = 0; s2 < 4; ++s2) Lg += __expf(sM[ct][s2][ln] - Mg) * sL[ct][s2][ln];
    const float fsc = __expf(mc - Mg);
    if (ssp == 0 && lh == 0) sInvL[ct * 16 + ln] = 1.0f / Lg;

    float rf[8];
    #pragma unroll
    for (int v = 0; v < 8; ++v) rf[v] = __shfl(fsc, v + 8 * lh, 32);
    #pragma unroll
    for (int dt = 0; dt < 16; ++dt)
      #pragma unroll
      for (int v = 0; v < 8; ++v)
        atomicAdd(&sComb[(ct * 16 + v + 8 * lh) * 256 + dt * 16 + ln], acc[dt][v] * rf[v]);
  }
  __syncthreads();

  // ---------- write context[b, c, d] ----------
  {
    const int row = tid >> 3;          // 0..31
    const int d0  = (tid & 7) * 32;
    const float inv = sInvL[row];
    float* dst = outg + ((size_t)batch * C_ + cblk + row) * D_ + d0;
    #pragma unroll
    for (int j = 0; j < 32; ++j) dst[j] = sComb[row * 256 + d0 + j] * inv;
  }
}

extern "C" void kernel_launch(void* const* d_in, const int* in_sizes, int n_in,
                              void* d_out, int out_size, void* d_ws, size_t ws_size,
                              hipStream_t stream) {
  (void)in_sizes; (void)n_in; (void)out_size; (void)d_ws; (void)ws_size;
  const float* x  = (const float*)d_in[0];
  const float* W  = (const float*)d_in[1];
  const float* bb = (const float*)d_in[2];
  float* out = (float*)d_out;
  dim3 grid(C_ / 32, B_);
  dim3 block(256);
  hipLaunchKernelGGL(elwa_kernel, grid, block, 0, stream, x, W, bb, out);
}
